// MultiModalSCGSnn_11828339933764
// MI455X (gfx1250) — matrix operand order repaired and checked by reading, not compiled, not run on hardware
//
#include <hip/hip_runtime.h>

// SNN: I1 = xf@W1^T (4096x1024x8192 GEMM), then 32-step LIF recurrence with
// s1@W2^T (4096x1024x64 per step) — both on v_wmma_f32_16x16x32_f16.
// W1/W2 pre-converted to f16 in workspace; GEMM K-loop is software-pipelined
// (iteration k+1 loads issued before iteration k's WMMAs).

typedef __attribute__((ext_vector_type(16))) _Float16 v16h;
typedef __attribute__((ext_vector_type(8)))  _Float16 v8h;
typedef __attribute__((ext_vector_type(8)))  float    v8f;

#define SNN_BETA 0.9f

__device__ __forceinline__ v8f wmma_f16(v16h a, v16h b, v8f c) {
  // D = A(16x32,f16) * B(32x16,f16) + C(16x16,f32)
  return __builtin_amdgcn_wmma_f32_16x16x32_f16(
      /*neg_a=*/false, a, /*neg_b=*/false, b,
      /*c_mod=*/(short)0, c, /*reuse_a=*/false, /*reuse_b=*/false);
}

__device__ __forceinline__ v16h pack16(const float4& x0, const float4& x1,
                                       const float4& x2, const float4& x3) {
  v16h r;
  r[0]=(_Float16)x0.x;  r[1]=(_Float16)x0.y;  r[2]=(_Float16)x0.z;  r[3]=(_Float16)x0.w;
  r[4]=(_Float16)x1.x;  r[5]=(_Float16)x1.y;  r[6]=(_Float16)x1.z;  r[7]=(_Float16)x1.w;
  r[8]=(_Float16)x2.x;  r[9]=(_Float16)x2.y;  r[10]=(_Float16)x2.z; r[11]=(_Float16)x2.w;
  r[12]=(_Float16)x3.x; r[13]=(_Float16)x3.y; r[14]=(_Float16)x3.z; r[15]=(_Float16)x3.w;
  return r;
}

__device__ __forceinline__ v16h load16h(const _Float16* p) {
  v8h lo = *(const v8h*)(p);
  v8h hi = *(const v8h*)(p + 16);
  return __builtin_shufflevector(lo, hi, 0,1,2,3,4,5,6,7,8,9,10,11,12,13,14,15);
}

// ---------------------------------------------------------------------------
// Kernel 0a: W1 (1024x8192 f32) -> f16 (vectorized: 8 elems/thread).
// ---------------------------------------------------------------------------
__global__ __launch_bounds__(256)
void snn_cvt_w1(const float* __restrict__ W1, _Float16* __restrict__ W1h) {
  size_t i = ((size_t)blockIdx.x * 256 + threadIdx.x) * 8;   // 8.39M elems total
  float4 x0 = *(const float4*)(W1 + i);
  float4 x1 = *(const float4*)(W1 + i + 4);
  v8h h;
  h[0]=(_Float16)x0.x; h[1]=(_Float16)x0.y; h[2]=(_Float16)x0.z; h[3]=(_Float16)x0.w;
  h[4]=(_Float16)x1.x; h[5]=(_Float16)x1.y; h[6]=(_Float16)x1.z; h[7]=(_Float16)x1.w;
  *(v8h*)(W1h + i) = h;
}

// ---------------------------------------------------------------------------
// Kernel 0b: W2 (64x1024 f32) -> f16 copy in workspace (stays hot in L2).
// ---------------------------------------------------------------------------
__global__ __launch_bounds__(256)
void snn_cvt_w2(const float* __restrict__ W2, _Float16* __restrict__ W2h) {
  int i = blockIdx.x * 256 + threadIdx.x;
  if (i < 64 * 1024) W2h[i] = (_Float16)W2[i];
}

// ---------------------------------------------------------------------------
// Kernel 1: I1 = xf @ W1h^T.  M=4096, N=1024, K=8192.
// Block: 256 thr = 8 waves as 2(M) x 4(N); each wave -> 16x64 output tile.
// A lane layout (16x32 f16): lane<16: M=lane, K runs {0..7,16..23}; lane>=16: +8.
// B lane layout (32x16 f16): lane<16: N=lane, K runs {0..7,16..23}; lane>=16: +8.
// Software-pipelined: next iteration's 12 loads are in flight across the
// current iteration's 4 WMMAs + 8 pk-cvts (double-buffered operand regs).
// ---------------------------------------------------------------------------
__global__ __launch_bounds__(256, 1)
void snn_fc1_gemm(const float* __restrict__ X, const _Float16* __restrict__ W1h,
                  float* __restrict__ I1) {
  const int lane  = threadIdx.x & 31;
  const int wave  = threadIdx.x >> 5;
  const int waveM = wave >> 2;           // 0..1
  const int waveN = wave & 3;            // 0..3
  const int mlane = lane & 15;
  const int hiA   = (lane >= 16) ? 8  : 0;   // A K-run offset
  const int hiB   = (lane >= 16) ? 8  : 0;   // B K-run offset

  const int m0 = blockIdx.x * 32 + waveM * 16;
  const int n0 = blockIdx.y * 256 + waveN * 64;

  const float*    pA  = X   + (size_t)(m0 + mlane) * 8192 + hiA;
  const _Float16* pB0 = W1h + (size_t)(n0 +  0 + mlane) * 8192 + hiB;
  const _Float16* pB1 = W1h + (size_t)(n0 + 16 + mlane) * 8192 + hiB;
  const _Float16* pB2 = W1h + (size_t)(n0 + 32 + mlane) * 8192 + hiB;
  const _Float16* pB3 = W1h + (size_t)(n0 + 48 + mlane) * 8192 + hiB;

  v8f acc0 = {}, acc1 = {}, acc2 = {}, acc3 = {};

  // --- prologue: operands for k = 0 ---
  float4 a0 = *(const float4*)(pA + 0);
  float4 a1 = *(const float4*)(pA + 4);
  float4 a2 = *(const float4*)(pA + 16);
  float4 a3 = *(const float4*)(pA + 20);
  v16h b0 = load16h(pB0);
  v16h b1 = load16h(pB1);
  v16h b2 = load16h(pB2);
  v16h b3 = load16h(pB3);
  v16h a  = pack16(a0, a1, a2, a3);

#pragma unroll 2
  for (int k = 0; k < 8192 - 32; k += 32) {
    const int kn = k + 32;
    // Prefetch ahead on both streams (pull toward near caches).
    __builtin_prefetch(pA + kn + 512, 0, 3);
    __builtin_prefetch(pB0 + kn + 1024, 0, 3);

    // Issue next iteration's loads first ...
    float4 na0 = *(const float4*)(pA + kn);
    float4 na1 = *(const float4*)(pA + kn + 4);
    float4 na2 = *(const float4*)(pA + kn + 16);
    float4 na3 = *(const float4*)(pA + kn + 20);
    v16h nb0 = load16h(pB0 + kn);
    v16h nb1 = load16h(pB1 + kn);
    v16h nb2 = load16h(pB2 + kn);
    v16h nb3 = load16h(pB3 + kn);

    // ... then compute on the current (already-resident) operands.
    acc0 = wmma_f16(a, b0, acc0);
    acc1 = wmma_f16(a, b1, acc1);
    acc2 = wmma_f16(a, b2, acc2);
    acc3 = wmma_f16(a, b3, acc3);

    a  = pack16(na0, na1, na2, na3);   // cvt after the WMMAs consumed old a
    b0 = nb0; b1 = nb1; b2 = nb2; b3 = nb3;
  }

  // --- epilogue: last K-step ---
  acc0 = wmma_f16(a, b0, acc0);
  acc1 = wmma_f16(a, b1, acc1);
  acc2 = wmma_f16(a, b2, acc2);
  acc3 = wmma_f16(a, b3, acc3);

  // C/D layout: VGPR r -> M = r + (lane>=16 ? 8 : 0), N = lane%16
  const int mc = (lane >= 16) ? 8 : 0;
#pragma unroll
  for (int r = 0; r < 8; ++r) {
    size_t m = (size_t)(m0 + r + mc);
    I1[m * 1024 + n0 +  0 + mlane] = acc0[r];
    I1[m * 1024 + n0 + 16 + mlane] = acc1[r];
    I1[m * 1024 + n0 + 32 + mlane] = acc2[r];
    I1[m * 1024 + n0 + 48 + mlane] = acc3[r];
  }
}

// ---------------------------------------------------------------------------
// Kernel 2: 32-step LIF recurrence. One block = 16 batch rows, 256 threads.
// v1/I1 slice in registers (64+64 f32/lane); spikes -> 32KB LDS f16 tile;
// waves 0..3 each own a 16x16 v2 tile: v2 = beta*v2 + s1 @ W2h^T via 32
// chained WMMAs; threshold/count is pure per-lane VALU on the C layout.
// ---------------------------------------------------------------------------
__global__ __launch_bounds__(256, 1)
void snn_recurrent(const float* __restrict__ I1, const _Float16* __restrict__ W2h,
                   float* __restrict__ out) {
  __shared__ _Float16 s1[16][1024];   // 32 KB spike tile, rewritten each step

  const int t     = threadIdx.x;
  const int lane  = t & 31;
  const int wave  = t >> 5;
  const int row   = t >> 4;           // 0..15 : batch row owned for v1 update
  const int c0    = t & 15;           // column phase (64 cols, stride 16)
  const int bbase = blockIdx.x * 16;
  const int mlane = lane & 15;
  const int hiA   = (lane >= 16) ? 8  : 0;
  const int hiB   = (lane >= 16) ? 8  : 0;
  const int hiC   = (lane >= 16) ? 8  : 0;

  float i1r[64], v1r[64];
  const float* pI = I1 + (size_t)(bbase + row) * 1024 + c0;
#pragma unroll
  for (int j = 0; j < 64; ++j) { i1r[j] = pI[16 * j]; v1r[j] = 0.0f; }

  const _Float16* pW = W2h + (size_t)((wave & 3) * 16 + mlane) * 1024 + hiB;
  const _Float16* pS = &s1[mlane][0] + hiA;
  _Float16* pSw = &s1[row][c0];

  v8f v2 = {}, cnt = {};

#pragma unroll 1
  for (int step = 0; step < 32; ++step) {
    // v1 = beta*v1 + I1 ; spike ; soft reset ; stash spike as f16 in LDS
#pragma unroll
    for (int j = 0; j < 64; ++j) {
      float v = SNN_BETA * v1r[j] + i1r[j];
      float s = (v >= 1.0f) ? 1.0f : 0.0f;
      v1r[j] = v - s;
      pSw[16 * j] = (_Float16)s;
    }
    __syncthreads();

    if (wave < 4) {              // uniform per-wave branch: EXEC stays all-ones
      v2 = v2 * SNN_BETA;        // leak folds into the WMMA accumulation chain
#pragma unroll 4
      for (int k = 0; k < 1024; k += 32) {
        v8h alo = *(const v8h*)(pS + k);        // A runs [k+off,+8)
        v8h ahi = *(const v8h*)(pS + k + 16);   //        [k+off+16,+8)
        v16h a = __builtin_shufflevector(alo, ahi,
                 0,1,2,3,4,5,6,7,8,9,10,11,12,13,14,15);
        v16h b = load16h(pW + k);               // B: same K runs as A
        v2 = wmma_f16(a, b, v2);
      }
#pragma unroll
      for (int r = 0; r < 8; ++r) {
        float s = (v2[r] >= 1.0f) ? 1.0f : 0.0f;
        v2[r] -= s;
        cnt[r] += s;
      }
    }
    __syncthreads();             // protect s1 before next step rewrites it
  }

  if (wave < 4) {
#pragma unroll
    for (int r = 0; r < 8; ++r) {
      size_t m = (size_t)(bbase + r + hiC);
      out[m * 64 + (wave & 3) * 16 + mlane] = cnt[r];
    }
  }
}

// ---------------------------------------------------------------------------
extern "C" void kernel_launch(void* const* d_in, const int* in_sizes, int n_in,
                              void* d_out, int out_size, void* d_ws, size_t ws_size,
                              hipStream_t stream) {
  const float* x  = (const float*)d_in[0];   // (4096, 8, 1024) == (4096, 8192)
  const float* W1 = (const float*)d_in[1];   // (1024, 8192)
  const float* W2 = (const float*)d_in[2];   // (64, 1024)
  float* out = (float*)d_out;                // (4096, 64)

  // Workspace layout (~32.1 MB):
  //   [0, 16MB)        I1   (4096x1024 f32)
  //   [16MB, 32MB)     W1h  (1024x8192 f16)
  //   [32MB, +128KB)   W2h  (64x1024 f16)
  char* ws = (char*)d_ws;
  float*    I1  = (float*)ws;
  _Float16* W1h = (_Float16*)(ws + ((size_t)16 << 20));
  _Float16* W2h = (_Float16*)(ws + ((size_t)32 << 20));

  snn_cvt_w1   <<<dim3(4096),   dim3(256), 0, stream>>>(W1, W1h);
  snn_cvt_w2   <<<dim3(256),    dim3(256), 0, stream>>>(W2, W2h);
  snn_fc1_gemm <<<dim3(128, 4), dim3(256), 0, stream>>>(x, W1h, I1);
  snn_recurrent<<<dim3(256),    dim3(256), 0, stream>>>(I1, W2h, out);
}
